// MultiModalAttentionLayer_5540507812538
// MI455X (gfx1250) — compile-verified, matrix operand order checked by reading, not run on hardware
//
#include <hip/hip_runtime.h>

// ---------------------------------------------------------------------------
// MultiModalAttentionLayer for MI455X (gfx1250, wave32, WMMA).
// Pipeline: f32->f16 convert -> 10 WMMA GEMM projections -> qrel/krel tables
//        -> 2 flash-attention passes (WMMA scores + WMMA P·V, online softmax)
//        -> sum ctx -> WMMA output GEMM (f32 -> d_out).
// CDNA5 specifics: v_wmma_f32_16x16x32_f16 for all matrix math,
// global_load_async_to_lds_b128 (ASYNCcnt) for tile staging with a
// double-buffered GEMM mainloop, global_prefetch for GL2 warmup.
// Workspace use: ~77 MB of d_ws.
// ---------------------------------------------------------------------------

typedef __attribute__((ext_vector_type(16))) _Float16 v16h;
typedef __attribute__((ext_vector_type(8)))  _Float16 v8h;
typedef __attribute__((ext_vector_type(8)))  float    v8f;

// Exact parameter type of the async-LDS builtins (from the clang diagnostic):
// pointer to int __attribute__((vector_size(16))) in AS(1)/AS(3).
typedef int v4i_t __attribute__((vector_size(16)));
typedef __attribute__((address_space(1))) v4i_t gas_v4i;
typedef __attribute__((address_space(3))) v4i_t las_v4i;

namespace {
constexpr int B_ = 4;
constexpr int S_ = 512;
constexpr int E_ = 768;
constexpr int H_ = 12;
constexpr int R_ = 17;   // 2*MAXREL+1
}

static __device__ __forceinline__ v8h ld8h(const _Float16* p) {
  return *reinterpret_cast<const v8h*>(p);
}
static __device__ __forceinline__ v16h mk16(v8h a, v8h b) {
  union { v16h v; v8h h[2]; } u;
  u.h[0] = a; u.h[1] = b;
  return u.v;
}
static __device__ __forceinline__ v8f wmma_f16(v16h a, v16h b, v8f c) {
  return __builtin_amdgcn_wmma_f32_16x16x32_f16(false, a, false, b, (short)0, c,
                                                false, false);
}

// 16-byte global -> LDS copy. Prefer the CDNA5 async path (ASYNCcnt-tracked
// GLOBAL_LOAD_ASYNC_TO_LDS_B128); fall back to a synchronous VGPR round-trip.
#if __has_builtin(__builtin_amdgcn_global_load_async_to_lds_b128)
#define HAVE_ASYNC_LDS 1
#else
#define HAVE_ASYNC_LDS 0
#endif

static __device__ __forceinline__ void async_cp16(const _Float16* g,
                                                  _Float16* l) {
#if HAVE_ASYNC_LDS
  __builtin_amdgcn_global_load_async_to_lds_b128((gas_v4i*)g, (las_v4i*)l,
                                                 0, 0);
#else
  *reinterpret_cast<v8h*>(l) = ld8h(g);
#endif
}

static __device__ __forceinline__ void wait_async() {
#if HAVE_ASYNC_LDS
#if __has_builtin(__builtin_amdgcn_s_wait_asynccnt)
  __builtin_amdgcn_s_wait_asynccnt(0);
#else
  asm volatile("s_wait_asynccnt 0x0" ::: "memory");
#endif
#endif
}

// ---------------------------------------------------------------------------
// Elementwise helpers
// ---------------------------------------------------------------------------
__global__ void f32tof16_kernel(const float* __restrict__ in,
                                _Float16* __restrict__ out, int n) {
  int i = blockIdx.x * blockDim.x + threadIdx.x;
  if (i < n) out[i] = (_Float16)in[i];
}

__global__ void sumcvt_kernel(const float* __restrict__ a,
                              const float* __restrict__ b,
                              _Float16* __restrict__ out, int n) {
  int i = blockIdx.x * blockDim.x + threadIdx.x;
  if (i < n) out[i] = (_Float16)(a[i] + b[i]);
}

// rel-position projection tables:
//   out[b][s][h][j] = dot(X[b,s,h*64:(h+1)*64], rel_table[j,:])   (j in 0..16)
__global__ void relproj_kernel(const _Float16* __restrict__ X,
                               const float* __restrict__ rel_table,
                               float* __restrict__ out, int total) {
  int idx = blockIdx.x * blockDim.x + threadIdx.x;
  if (idx >= total) return;
  int j = idx % R_;
  int h = (idx / R_) % H_;
  int s = (idx / (R_ * H_)) % S_;
  int b = idx / (R_ * H_ * S_);
  const _Float16* x = X + (((size_t)b * S_ + s) * E_) + h * 64;
  const float* rt = rel_table + j * 64;
  float acc = 0.f;
#pragma unroll
  for (int d = 0; d < 64; ++d) acc += (float)x[d] * rt[d];
  out[idx] = acc;
}

// ---------------------------------------------------------------------------
// GEMM: C[M,N] = A[M,K] * W[N,K]^T + bias[N]      (f16 in, f32 accum)
// Block tile 128x128, K step 32, 8 waves; wave tile 64x32 (8 accum tiles).
// Double-buffered LDS with async global->LDS staging: tile t+1 is in flight
// while WMMAs consume tile t. End-of-iteration barrier makes the 2-buffer
// rotation safe (the buffer being filled was last read in the previous iteration).
// ---------------------------------------------------------------------------
template <bool STORE_F16>
__global__ __launch_bounds__(256) void gemm_bias_kernel(
    const _Float16* __restrict__ A, const _Float16* __restrict__ W,
    const float* __restrict__ bias, void* __restrict__ Cout,
    int M, int N, int K) {
  __shared__ __align__(16) _Float16 sA[2][128 * 32];
  __shared__ __align__(16) _Float16 sW[2][128 * 32];

  const int tid = threadIdx.x;
  const int lane = tid & 31;
  const int c = lane & 15;        // N (or M) index within 16-lane group
  const int hi = lane >> 4;       // lane-group select (K sub-block)
  const int w = tid >> 5;         // wave 0..7
  const int wm = w & 1;           // 2 waves along M
  const int wn = w >> 1;          // 4 waves along N
  const int bm = blockIdx.y * 128;
  const int bn = blockIdx.x * 128;

  v8f acc[4][2];
#pragma unroll
  for (int i = 0; i < 4; ++i)
#pragma unroll
    for (int j = 0; j < 2; ++j) acc[i][j] = v8f{};

  const int lr = tid >> 1;        // 0..127 : row loaded by this thread
  const int lk = (tid & 1) * 16;  // 0 / 16 : K half loaded by this thread

  auto stage = [&](int buf, int kk) {
    size_t ga = (size_t)(bm + lr) * K + kk + lk;
    size_t gw = (size_t)(bn + lr) * K + kk + lk;
    async_cp16(A + ga, &sA[buf][lr * 32 + lk]);
    async_cp16(A + ga + 8, &sA[buf][lr * 32 + lk + 8]);
    async_cp16(W + gw, &sW[buf][lr * 32 + lk]);
    async_cp16(W + gw + 8, &sW[buf][lr * 32 + lk + 8]);
    if (kk + 32 < K) {  // warm GL2 one tile ahead
      __builtin_prefetch(A + ga + 32, 0, 1);
      __builtin_prefetch(W + gw + 32, 0, 1);
    }
  };

  const int nk = K / 32;
  stage(0, 0);
  wait_async();
  __syncthreads();

  for (int t = 0; t < nk; ++t) {
    if (t + 1 < nk) stage((t + 1) & 1, (t + 1) * 32);

    const _Float16* cA = sA[t & 1];
    const _Float16* cW = sW[t & 1];
    v16h af[4];
#pragma unroll
    for (int i = 0; i < 4; ++i) {
      int row = wm * 64 + i * 16 + c;
      af[i] = mk16(ld8h(&cA[row * 32 + hi * 8]),
                   ld8h(&cA[row * 32 + 16 + hi * 8]));
    }
#pragma unroll
    for (int j = 0; j < 2; ++j) {
      int row = wn * 32 + j * 16 + c;
      v16h bf = mk16(ld8h(&cW[row * 32 + hi * 8]),
                     ld8h(&cW[row * 32 + 16 + hi * 8]));
#pragma unroll
      for (int i = 0; i < 4; ++i) acc[i][j] = wmma_f16(af[i], bf, acc[i][j]);
    }

    wait_async();
    __syncthreads();
  }

#pragma unroll
  for (int j = 0; j < 2; ++j) {
    int n = bn + wn * 32 + j * 16 + c;
    float bv = bias[n];
#pragma unroll
    for (int i = 0; i < 4; ++i) {
#pragma unroll
      for (int r = 0; r < 8; ++r) {
        int m = bm + wm * 64 + i * 16 + r + 8 * hi;  // C layout: M = r + 8*hi
        float v = acc[i][j][r] + bv;
        if (STORE_F16)
          ((_Float16*)Cout)[(size_t)m * N + n] = (_Float16)v;
        else
          ((float*)Cout)[(size_t)m * N + n] = v;
      }
    }
  }
}

// ---------------------------------------------------------------------------
// Flash attention per modality, per (b,h), 128 queries / workgroup.
// scores = (q·k^T + qs·ks^T)/sqrt(E) + krel[t,idx] + qrel[l,idx]
// idx = clip(t-l,-8,8)+8.  Online softmax, then P·V, write ctx (f32).
// K/KS tiles staged with async global->LDS; V staged manually (transposed).
// ---------------------------------------------------------------------------
__global__ __launch_bounds__(256) void attn_kernel(
    const _Float16* __restrict__ Q, const _Float16* __restrict__ K,
    const _Float16* __restrict__ V, const _Float16* __restrict__ QS,
    const _Float16* __restrict__ KS, const float* __restrict__ qrel,
    const float* __restrict__ krel, float* __restrict__ ctx) {
  constexpr float inv_scale = 0.03608439182435161f;  // 1/sqrt(768)

  const int tid = threadIdx.x;
  const int lane = tid & 31;
  const int c = lane & 15;
  const int hi = lane >> 4;
  const int w = tid >> 5;                 // wave 0..7 -> 16 queries each
  const int b = blockIdx.y / H_;
  const int h = blockIdx.y % H_;
  const int qbase = blockIdx.x * 128;
  const int q0 = qbase + w * 16;

  __shared__ __align__(16) _Float16 sK[32 * 64];    // [key][dim]
  __shared__ __align__(16) _Float16 sKS[32 * 64];   // [key][dim]
  __shared__ __align__(16) _Float16 sVT[64 * 32];   // [dim][key] (transposed)
  __shared__ __align__(16) _Float16 sP[8][16 * 32]; // per-wave P scratch
  __shared__ float sKrel[32 * R_];
  __shared__ float sQrel[128 * R_];

  // qrel table for this query block (persistent over key loop)
  for (int i = tid; i < 128 * R_; i += 256) {
    int l = i / R_, j = i % R_;
    sQrel[i] = qrel[(((size_t)b * S_ + qbase + l) * H_ + h) * R_ + j];
  }

  // A-fragments for q and qs: 16 queries x 64 dims -> 2 frags (K=32 each)
  const size_t rowQ = (((size_t)b * S_ + q0 + c) * E_) + h * 64;
  v16h aq[2], asp[2];
#pragma unroll
  for (int f = 0; f < 2; ++f) {
    aq[f]  = mk16(ld8h(Q + rowQ + f * 32 + hi * 8),
                  ld8h(Q + rowQ + f * 32 + 16 + hi * 8));
    asp[f] = mk16(ld8h(QS + rowQ + f * 32 + hi * 8),
                  ld8h(QS + rowQ + f * 32 + 16 + hi * 8));
  }

  v8f o[4];
#pragma unroll
  for (int jd = 0; jd < 4; ++jd) o[jd] = v8f{};
  float mrow[8], lrow[8];
#pragma unroll
  for (int r = 0; r < 8; ++r) { mrow[r] = -1e30f; lrow[r] = 0.f; }

  for (int it = 0; it < S_ / 32; ++it) {
    const int kb = it * 32;
    __syncthreads();  // protect LDS tiles from previous iteration
    {
      // stage K, KS (async, verbatim copy) and V transposed (manual)
      int key = tid >> 3;
      int d0 = (tid & 7) * 8;
      size_t g = (((size_t)b * S_ + kb + key) * E_) + h * 64 + d0;
      async_cp16(K + g, &sK[key * 64 + d0]);
      async_cp16(KS + g, &sKS[key * 64 + d0]);
      v8h vv = ld8h(V + g);
#pragma unroll
      for (int j = 0; j < 8; ++j) sVT[(d0 + j) * 32 + key] = vv[j];
    }
    for (int i = tid; i < 32 * R_; i += 256) {
      int t = i / R_, j = i % R_;
      sKrel[i] = krel[(((size_t)b * S_ + kb + t) * H_ + h) * R_ + j];
    }
    wait_async();
    __syncthreads();

    // ---- scores: two 16x16 tiles covering 32 keys ----
    float sv[2][8];
#pragma unroll
    for (int j = 0; j < 2; ++j) {
      int rowk = (j * 16 + c) * 64;
      v16h bk0 = mk16(ld8h(&sK[rowk + hi * 8]), ld8h(&sK[rowk + 16 + hi * 8]));
      v16h bk1 = mk16(ld8h(&sK[rowk + 32 + hi * 8]), ld8h(&sK[rowk + 48 + hi * 8]));
      v16h bs0 = mk16(ld8h(&sKS[rowk + hi * 8]), ld8h(&sKS[rowk + 16 + hi * 8]));
      v16h bs1 = mk16(ld8h(&sKS[rowk + 32 + hi * 8]), ld8h(&sKS[rowk + 48 + hi * 8]));
      v8f acc = v8f{};
      acc = wmma_f16(aq[0], bk0, acc);
      acc = wmma_f16(aq[1], bk1, acc);
      acc = wmma_f16(asp[0], bs0, acc);
      acc = wmma_f16(asp[1], bs1, acc);
#pragma unroll
      for (int r = 0; r < 8; ++r) {
        int t = kb + j * 16 + c;
        int l = q0 + r + 8 * hi;
        int di = t - l;
        di = di < -8 ? -8 : (di > 8 ? 8 : di);
        int ri = di + 8;
        sv[j][r] = acc[r] * inv_scale + sKrel[(j * 16 + c) * R_ + ri] +
                   sQrel[(w * 16 + r + 8 * hi) * R_ + ri];
      }
    }

    // ---- online softmax over the 32-key strip ----
#pragma unroll
    for (int r = 0; r < 8; ++r) {
      float vmax = fmaxf(sv[0][r], sv[1][r]);
      vmax = fmaxf(vmax, __shfl_xor(vmax, 1, 32));
      vmax = fmaxf(vmax, __shfl_xor(vmax, 2, 32));
      vmax = fmaxf(vmax, __shfl_xor(vmax, 4, 32));
      vmax = fmaxf(vmax, __shfl_xor(vmax, 8, 32));
      float mnew = fmaxf(mrow[r], vmax);
      float alpha = __expf(mrow[r] - mnew);
      float p0 = __expf(sv[0][r] - mnew);
      float p1 = __expf(sv[1][r] - mnew);
      float ps = p0 + p1;
      ps += __shfl_xor(ps, 1, 32);
      ps += __shfl_xor(ps, 2, 32);
      ps += __shfl_xor(ps, 4, 32);
      ps += __shfl_xor(ps, 8, 32);
      lrow[r] = lrow[r] * alpha + ps;
      mrow[r] = mnew;
#pragma unroll
      for (int jd = 0; jd < 4; ++jd) o[jd][r] *= alpha;
      int prow = (r + 8 * hi) * 32;
      sP[w][prow + c] = (_Float16)p0;
      sP[w][prow + 16 + c] = (_Float16)p1;
    }

    // ---- P·V via LDS round-trip (same-wave DS ops are in-order) ----
    {
      const _Float16* pb = sP[w];
      v16h ap = mk16(ld8h(&pb[c * 32 + hi * 8]), ld8h(&pb[c * 32 + 16 + hi * 8]));
#pragma unroll
      for (int jd = 0; jd < 4; ++jd) {
        int rowv = (jd * 16 + c) * 32;
        v16h bv = mk16(ld8h(&sVT[rowv + hi * 8]),
                       ld8h(&sVT[rowv + 16 + hi * 8]));
        o[jd] = wmma_f16(ap, bv, o[jd]);
      }
    }
  }

  // ---- normalize and write ctx ----
#pragma unroll
  for (int jd = 0; jd < 4; ++jd) {
#pragma unroll
    for (int r = 0; r < 8; ++r) {
      int row = q0 + r + 8 * hi;
      int dd = jd * 16 + c;
      ctx[(((size_t)b * S_ + row) * E_) + h * 64 + dd] = o[jd][r] / lrow[r];
    }
  }
}

// ---------------------------------------------------------------------------
// Host orchestration
// ---------------------------------------------------------------------------
extern "C" void kernel_launch(void* const* d_in, const int* in_sizes, int n_in,
                              void* d_out, int out_size, void* d_ws,
                              size_t ws_size, hipStream_t stream) {
  (void)in_sizes; (void)n_in; (void)out_size; (void)ws_size;

  constexpr size_t MNe = (size_t)B_ * S_ * E_;        // 1,572,864
  constexpr size_t WNe = (size_t)E_ * E_;             //   589,824
  constexpr size_t RELe = (size_t)B_ * S_ * H_ * R_;  //   417,792

  size_t off = 0;
  auto carve = [&](size_t bytes) -> void* {
    void* p = (char*)d_ws + off;
    off += (bytes + 255) & ~(size_t)255;
    return p;
  };

  _Float16* fx[4];
  for (int i = 0; i < 4; ++i) fx[i] = (_Float16*)carve(MNe * 2);
  _Float16* fw[9];
  for (int i = 0; i < 9; ++i) fw[i] = (_Float16*)carve(WNe * 2);
  _Float16* pj[10];
  for (int i = 0; i < 10; ++i) pj[i] = (_Float16*)carve(MNe * 2);
  float* qrel_t = (float*)carve(RELe * 4);
  float* krel_t = (float*)carve(RELe * 4);
  float* qrel_i = (float*)carve(RELe * 4);
  float* krel_i = (float*)carve(RELe * 4);
  float* ctx_t = (float*)carve(MNe * 4);
  float* ctx_i = (float*)carve(MNe * 4);
  _Float16* csum = (_Float16*)carve(MNe * 2);

  // 1) convert features + weights to f16
  for (int i = 0; i < 4; ++i)
    f32tof16_kernel<<<(int)((MNe + 255) / 256), 256, 0, stream>>>(
        (const float*)d_in[i], fx[i], (int)MNe);
  const int widx[9] = {4, 6, 8, 10, 12, 14, 16, 18, 20};  // Wq_t..Wk_s, Wo
  for (int i = 0; i < 9; ++i)
    f32tof16_kernel<<<(int)((WNe + 255) / 256), 256, 0, stream>>>(
        (const float*)d_in[widx[i]], fw[i], (int)WNe);

  // 2) ten projection GEMMs (f16 output)
  dim3 gg(E_ / 128, (B_ * S_) / 128);  // (6,16)
  const int M = B_ * S_;
  // text: q,k,v ; text_spatial: qs,ks
  gemm_bias_kernel<true><<<gg, 256, 0, stream>>>(fx[0], fw[0], (const float*)d_in[5],  pj[0], M, E_, E_);
  gemm_bias_kernel<true><<<gg, 256, 0, stream>>>(fx[0], fw[1], (const float*)d_in[7],  pj[1], M, E_, E_);
  gemm_bias_kernel<true><<<gg, 256, 0, stream>>>(fx[0], fw[2], (const float*)d_in[9],  pj[2], M, E_, E_);
  gemm_bias_kernel<true><<<gg, 256, 0, stream>>>(fx[2], fw[6], (const float*)d_in[17], pj[3], M, E_, E_);
  gemm_bias_kernel<true><<<gg, 256, 0, stream>>>(fx[2], fw[7], (const float*)d_in[19], pj[4], M, E_, E_);
  // img: q,k,v ; img_spatial: qs,ks
  gemm_bias_kernel<true><<<gg, 256, 0, stream>>>(fx[1], fw[3], (const float*)d_in[11], pj[5], M, E_, E_);
  gemm_bias_kernel<true><<<gg, 256, 0, stream>>>(fx[1], fw[4], (const float*)d_in[13], pj[6], M, E_, E_);
  gemm_bias_kernel<true><<<gg, 256, 0, stream>>>(fx[1], fw[5], (const float*)d_in[15], pj[7], M, E_, E_);
  gemm_bias_kernel<true><<<gg, 256, 0, stream>>>(fx[3], fw[6], (const float*)d_in[17], pj[8], M, E_, E_);
  gemm_bias_kernel<true><<<gg, 256, 0, stream>>>(fx[3], fw[7], (const float*)d_in[19], pj[9], M, E_, E_);

  // 3) rel-position tables from q and k of each modality
  const float* rel_table = (const float*)d_in[22];
  int rgrid = (int)((RELe + 255) / 256);
  relproj_kernel<<<rgrid, 256, 0, stream>>>(pj[0], rel_table, qrel_t, (int)RELe);
  relproj_kernel<<<rgrid, 256, 0, stream>>>(pj[1], rel_table, krel_t, (int)RELe);
  relproj_kernel<<<rgrid, 256, 0, stream>>>(pj[5], rel_table, qrel_i, (int)RELe);
  relproj_kernel<<<rgrid, 256, 0, stream>>>(pj[6], rel_table, krel_i, (int)RELe);

  // 4) flash attention per modality
  dim3 ga(S_ / 128, B_ * H_);  // (4,48)
  attn_kernel<<<ga, 256, 0, stream>>>(pj[0], pj[1], pj[2], pj[3], pj[4],
                                      qrel_t, krel_t, ctx_t);
  attn_kernel<<<ga, 256, 0, stream>>>(pj[5], pj[6], pj[7], pj[8], pj[9],
                                      qrel_i, krel_i, ctx_i);

  // 5) ctx_t + ctx_i -> f16, then output projection (f32 -> d_out)
  sumcvt_kernel<<<(int)((MNe + 255) / 256), 256, 0, stream>>>(
      ctx_t, ctx_i, csum, (int)MNe);
  gemm_bias_kernel<false><<<gg, 256, 0, stream>>>(
      csum, fw[8], (const float*)d_in[21], d_out, M, E_, E_);
}
